// ToepGreenMgNet_29686813950814
// MI455X (gfx1250) — matrix operand order repaired and checked by reading, not compiled, not run on hardware
//
#include <hip/hip_runtime.h>
#include <stdint.h>

// ---------------------------------------------------------------------------
// ToepGreenMgNet on MI455X (gfx1250):
//   a = multigrid-MLP kernel (shared across batch, length 16385)
//   w[b,i] = sum_j a[8192 + i - j] * u[b,j]   (Toeplitz GEMM, bf16 WMMA)
// ---------------------------------------------------------------------------

typedef __attribute__((ext_vector_type(16))) unsigned short v16u;
typedef __attribute__((ext_vector_type(16))) __bf16         v16bf;
typedef __attribute__((ext_vector_type(8)))  float          v8f;
typedef __attribute__((ext_vector_type(4)))  unsigned int   v4u;
typedef __attribute__((ext_vector_type(8)))  int            v8i;
typedef __attribute__((ext_vector_type(4)))  int            v4i;

#define BATCH   1024
#define NPTS    8193            // N
#define LFULL   16385           // 2N-1
#define NC      2049            // coarse grid points
#define KPAD    8224            // 257 * 32 (zero-padded K)
#define KSTEPS  257
#define BK      32
#define TILE_I  256             // i-tile per workgroup
#define TILE_B  64              // batch-tile per workgroup

// workspace layout (bytes); total ~17 MB
#define WS_UBF_OFF    0ull
#define WS_UBF_BYTES  ((unsigned long long)BATCH * KPAD * 2ull)
#define ABF_LEN       16672     // 31 zero-pad + 16385 + tail zero-pad
#define WS_ABF_OFF    (WS_UBF_OFF + WS_UBF_BYTES)
#define WS_BANDS_OFF  (WS_ABF_OFF + (unsigned long long)ABF_LEN * 2ull)
#define WS_A1_OFF     (WS_BANDS_OFF + 3ull * NC * 4ull)
#define WS_A2_OFF     (WS_A1_OFF + 4100ull * 4ull)

#if defined(__has_builtin)
#if __has_builtin(__builtin_amdgcn_tensor_load_to_lds) && __has_builtin(__builtin_amdgcn_s_wait_tensorcnt)
#define HAVE_TDM 1
#else
#define HAVE_TDM 0
#endif
#else
#define HAVE_TDM 0
#endif

__device__ __forceinline__ unsigned short f2bf(float f) {
  unsigned u = __float_as_uint(f);
  u += 0x7FFFu + ((u >> 16) & 1u);   // round-to-nearest-even
  return (unsigned short)(u >> 16);
}

// ---------------------------------------------------------------------------
// 1) u (f32, B x 8193) -> bf16, zero-padded to B x 8224
// ---------------------------------------------------------------------------
__global__ void convert_u_kernel(const float* __restrict__ u,
                                 unsigned short* __restrict__ ubf) {
  long long idx = (long long)blockIdx.x * blockDim.x + threadIdx.x;
  const long long total = (long long)BATCH * KPAD;
  if (idx >= total) return;
  const int b = (int)(idx / KPAD);
  const int k = (int)(idx % KPAD);
  const float v = (k < NPTS) ? u[(long long)b * NPTS + k] : 0.0f;
  ubf[idx] = f2bf(v);
}

// ---------------------------------------------------------------------------
// 2) per-level pointwise MLP bands on the 2049-point coarse grid
// ---------------------------------------------------------------------------
__global__ void mlp_bands_kernel(const float* __restrict__ x,
                                 const float* __restrict__ W1, const float* __restrict__ b1,
                                 const float* __restrict__ W2, const float* __restrict__ b2,
                                 const float* __restrict__ W3, const float* __restrict__ b3,
                                 float* __restrict__ bands) {
  const int j = blockIdx.y;                                  // level 0..2
  const int p = blockIdx.x * blockDim.x + threadIdx.x;       // coarse point
  __shared__ float sW2[64 * 64];
  __shared__ float sW1[64], sb1[64], sb2[64], sW3[64];
  for (int t = threadIdx.x; t < 64 * 64; t += blockDim.x) sW2[t] = W2[j * 4096 + t];
  if (threadIdx.x < 64) {
    sW1[threadIdx.x] = W1[j * 64 + threadIdx.x];
    sb1[threadIdx.x] = b1[j * 64 + threadIdx.x];
    sb2[threadIdx.x] = b2[j * 64 + threadIdx.x];
    sW3[threadIdx.x] = W3[j * 64 + threadIdx.x];
  }
  __syncthreads();
  if (p >= NC) return;
  const float xc = x[8 * p];   // x row 0: injection1d applied 3x == every 8th point
  float h1[64];
#pragma unroll
  for (int c = 0; c < 64; ++c) h1[c] = tanhf(sW1[c] * xc + sb1[c]);
  float out = b3[j];
  for (int o = 0; o < 64; ++o) {
    float acc = sb2[o];
#pragma unroll
    for (int c = 0; c < 64; ++c) acc += sW2[o * 64 + c] * h1[c];
    out += sW3[o] * tanhf(acc);
  }
  bands[j * NC + p] = out;
}

// ---------------------------------------------------------------------------
// 3) splice + interp chain; emit zero-padded bf16 a  (abf[31 + g] = a[g])
// ---------------------------------------------------------------------------
__global__ void assemble_a_kernel(const float* __restrict__ bands,
                                  float* __restrict__ a1, float* __restrict__ a2,
                                  unsigned short* __restrict__ abf) {
  const int tid = threadIdx.x, nt = blockDim.x;
  const float* band0 = bands;
  const float* band1 = bands + NC;
  const float* band2 = bands + 2 * NC;
  for (int t = tid; t < 4097; t += nt)
    a1[t] = (t & 1) ? 0.5f * (band0[t >> 1] + band0[(t >> 1) + 1]) : band0[t >> 1];
  __threadfence_block(); __syncthreads();
  for (int t = tid; t < NC; t += nt) a1[512 + t] = band1[t];
  __threadfence_block(); __syncthreads();
  for (int t = tid; t < 8193; t += nt)
    a2[t] = (t & 1) ? 0.5f * (a1[t >> 1] + a1[(t >> 1) + 1]) : a1[t >> 1];
  __threadfence_block(); __syncthreads();
  for (int t = tid; t < NC; t += nt) a2[1024 + t] = band2[t];
  __threadfence_block(); __syncthreads();
  for (int t = tid; t < ABF_LEN; t += nt) {
    unsigned short v = 0;
    const int g = t - 31;
    if (g >= 0 && g < LFULL) {
      const float av = (g & 1) ? 0.5f * (a2[g >> 1] + a2[(g >> 1) + 1]) : a2[g >> 1];
      v = f2bf(av);
    }
    abf[t] = v;
  }
}

// ---------------------------------------------------------------------------
// 4) Toeplitz GEMM via v_wmma_f32_16x16x32_bf16, TDM-fed LDS double buffer
// ---------------------------------------------------------------------------
#define ASLICE_LEN 8479
#define SM_ABYTES  16960                 // 8480 halves
#define SM_UOFF    SM_ABYTES
#define SM_UBYTES  (TILE_B * BK * 2)     // 4096 per buffer

union BF16V { v16u u; v16bf b; };

#if HAVE_TDM
__device__ __forceinline__ void tdm_issue(unsigned lds_off, const unsigned short* gsrc) {
  // D# per CDNA5 ISA ch.8: group0 = {flags, lds_addr, global_addr, type=2}
  const unsigned long long ga = (unsigned long long)(uintptr_t)gsrc;
  v4u g0;
  g0.x = 1u;                                                   // count=1, user mode
  g0.y = lds_off;                                              // LDS byte address
  g0.z = (unsigned)ga;                                         // global_addr[31:0]
  g0.w = ((unsigned)(ga >> 32) & 0x01FFFFFFu) | 0x80000000u;   // addr[56:32] | type=2
  // group1: data_size=2B, tensor_dim0=8224, tensor_dim1=4096,
  //         tile_dim0=32, tile_dim1=64, tensor_dim0_stride=8224
  v8i g1 = { (int)0x00010000,
             (int)(8224u << 16),
             (int)(4096u << 16),
             (int)(32u << 16),
             64,
             8224,
             0, 0 };
  v4i z4 = {0, 0, 0, 0};
  v8i z8 = {0, 0, 0, 0, 0, 0, 0, 0};
  // 6-arg form on this toolchain (clang-23 / therock-10.0 headers):
  // (group0, group1, group2, group3, ext, cpol)
  __builtin_amdgcn_tensor_load_to_lds(g0, g1, z4, z4, z8, 0);
}
#endif

__global__ void __launch_bounds__(256, 2)
toep_wmma_kernel(const unsigned short* __restrict__ ubf,
                 const unsigned short* __restrict__ abf,
                 float* __restrict__ out) {
  __shared__ alignas(16) unsigned char smem[SM_ABYTES + 2 * SM_UBYTES];
  unsigned short* asr = (unsigned short*)smem;   // reversed a-slice

  const int tid  = threadIdx.x;
  const int lane = tid & 31;
  const int hi   = lane >> 4;                    // lane half (wave32 WMMA layout)
  const int ln   = lane & 15;
  const int wave = tid >> 5;
  const int wrow = wave >> 1;                    // 0..3 : i direction, 64 rows each
  const int wcol = wave & 1;                     // 0..1 : batch direction, 32 each
  const int i0   = blockIdx.x * TILE_I;
  const int b0   = blockIdx.y * TILE_B;

#if HAVE_TDM
  if (wave == 0) tdm_issue(SM_UOFF, ubf + (long long)b0 * KPAD);   // tile k=0
#endif

  // Stage reversed Toeplitz slice: asr[s] = a[i0 + 8447 - s], zero-padded ends.
  for (int s = tid; s < ASLICE_LEN; s += 256)
    asr[s] = abf[31 + i0 + 8447 - s];

  v8f acc[4][2];
#pragma unroll
  for (int fa = 0; fa < 4; ++fa)
#pragma unroll
    for (int fb = 0; fb < 2; ++fb) acc[fa][fb] = (v8f)0.0f;

  for (int k = 0; k < KSTEPS; ++k) {
    const int j0 = k * BK;
#if HAVE_TDM
    __builtin_amdgcn_s_wait_tensorcnt(0);        // tile k landed in LDS
    __syncthreads();                             // all waves: tile k visible, k-1 done
    if (wave == 0 && k + 1 < KSTEPS)
      tdm_issue(SM_UOFF + (unsigned)(((k + 1) & 1) * SM_UBYTES),
                ubf + (long long)b0 * KPAD + (long long)(k + 1) * BK);
#else
    for (int t = tid; t < TILE_B * BK; t += 256) {
      const int bl = t >> 5, kk = t & 31;
      ((unsigned short*)(smem + SM_UOFF + (k & 1) * SM_UBYTES))[t] =
          ubf[(long long)(b0 + bl) * KPAD + j0 + kk];
    }
    __syncthreads();
#endif
    const unsigned short* ub = (const unsigned short*)(smem + SM_UOFF + (k & 1) * SM_UBYTES);

    // B fragments (u): 32x16 bf16, lane<16: col=lane,K=h; lane>=16: K=16+h.
    BF16V bf0, bf1;
    {
      const unsigned short* bp = ub + (wcol * 32 + ln) * BK + hi * 16;
      bf0.u = *(const v16u*)bp;             // batches  +0..15 of wave's 32
      bf1.u = *(const v16u*)(bp + 16 * BK); // batches +16..31
    }

    // A fragments (Toeplitz): 16x32 bf16, m=lane&15, K = h + 16*hi.
#pragma unroll
    for (int fa = 0; fa < 4; ++fa) {
      const int local_i = wrow * 64 + fa * 16 + ln;
      const int sbase   = 255 - local_i + j0 + (hi ? 16 : 0);  // reversed slice base
      BF16V av;
#pragma unroll
      for (int h = 0; h < 16; ++h) {
        const int Kh = h;
        av.u[h] = asr[sbase + Kh];
      }
      acc[fa][0] = __builtin_amdgcn_wmma_f32_16x16x32_bf16(
          false, av.b, false, bf0.b, (short)0, acc[fa][0], false, false);
      acc[fa][1] = __builtin_amdgcn_wmma_f32_16x16x32_bf16(
          false, av.b, false, bf1.b, (short)0, acc[fa][1], false, false);
    }
  }

  // D layout: lane<16: col=batch=lane, VGPR r -> M = r + 8*hi (i direction).
#pragma unroll
  for (int fa = 0; fa < 4; ++fa) {
    const int ibase = i0 + wrow * 64 + fa * 16 + hi * 8;
#pragma unroll
    for (int fb = 0; fb < 2; ++fb) {
      const int b = b0 + wcol * 32 + fb * 16 + ln;
      float* op = out + (long long)b * NPTS;
      const v8f c = acc[fa][fb];
#pragma unroll
      for (int r = 0; r < 8; ++r) {
        const int i = ibase + r;
        if (i < NPTS) op[i] = c[r];
      }
    }
  }
}

// ---------------------------------------------------------------------------
extern "C" void kernel_launch(void* const* d_in, const int* in_sizes, int n_in,
                              void* d_out, int out_size, void* d_ws, size_t ws_size,
                              hipStream_t stream) {
  const float* u  = (const float*)d_in[0];
  const float* x  = (const float*)d_in[1];
  const float* W1 = (const float*)d_in[2];
  const float* b1 = (const float*)d_in[3];
  const float* W2 = (const float*)d_in[4];
  const float* b2 = (const float*)d_in[5];
  const float* W3 = (const float*)d_in[6];
  const float* b3 = (const float*)d_in[7];
  float* out = (float*)d_out;

  unsigned char* ws = (unsigned char*)d_ws;     // needs ~17 MB
  unsigned short* ubf  = (unsigned short*)(ws + WS_UBF_OFF);
  unsigned short* abf  = (unsigned short*)(ws + WS_ABF_OFF);
  float* bands = (float*)(ws + WS_BANDS_OFF);
  float* a1    = (float*)(ws + WS_A1_OFF);
  float* a2    = (float*)(ws + WS_A2_OFF);

  {
    const long long total = (long long)BATCH * KPAD;
    const int blocks = (int)((total + 255) / 256);
    convert_u_kernel<<<blocks, 256, 0, stream>>>(u, ubf);
  }
  mlp_bands_kernel<<<dim3(9, 3), 256, 0, stream>>>(x, W1, b1, W2, b2, W3, b3, bands);
  assemble_a_kernel<<<1, 256, 0, stream>>>(bands, a1, a2, abf);
  toep_wmma_kernel<<<dim3((NPTS + TILE_I - 1) / TILE_I, BATCH / TILE_B), 256, 0, stream>>>(
      ubf, abf, out);
}
